// Encoder_54949811585416
// MI455X (gfx1250) — compile-verified, matrix operand order checked by reading, not run on hardware
//
#include <hip/hip_runtime.h>
#include <cmath>

constexpr int Bsz = 4096;
constexpr int Tsz = 128;
constexpr int Nsz = 128;
constexpr int Hsz = 128;
#define EPSf 1e-5f

typedef __bf16 v16bf __attribute__((ext_vector_type(16)));
typedef float  v8f   __attribute__((ext_vector_type(8)));

union BF16x16 { v16bf v; __bf16 e[16]; };

__device__ __forceinline__ float sigmoidf(float x) { return 1.0f / (1.0f + expf(-x)); }

// ---------------------------------------------------------------------------
// Prep 1: alpha[b,n] = softmax_n( sum_t X[b,t,n]*w_x[t] + attn_b )
// (the h1/c1 score terms are constant along the softmax axis and cancel)
// ---------------------------------------------------------------------------
__global__ void alpha_kernel(const float* __restrict__ X,
                             const float* __restrict__ attn_w,
                             const float* __restrict__ attn_b,
                             float* __restrict__ alpha) {
  const int b = blockIdx.x;
  const int n = threadIdx.x;
  const float* wx = attn_w + 2 * Hsz;
  const float* xp = X + (size_t)b * Tsz * Nsz + n;
  float s = 0.f;
  #pragma unroll 4
  for (int t = 0; t < Tsz; ++t) s += xp[(size_t)t * Nsz] * wx[t];
  s += attn_b[0];

  __shared__ float red[Nsz];
  red[n] = s; __syncthreads();
  for (int off = 64; off > 0; off >>= 1) {
    if (n < off) red[n] = fmaxf(red[n], red[n + off]);
    __syncthreads();
  }
  const float mx = red[0];
  __syncthreads();
  const float e = expf(s - mx);
  red[n] = e; __syncthreads();
  for (int off = 64; off > 0; off >>= 1) {
    if (n < off) red[n] += red[n + off];
    __syncthreads();
  }
  alpha[(size_t)b * Nsz + n] = e / red[0];
}

// ---------------------------------------------------------------------------
// Prep 2: pack the 4 weight matrices (4H x N, row-major f32) into bf16
// B-fragments laid out so each lane reads 16 contiguous bf16 (32B) per
// (jt, kc) tile.  B element (k,n) = W[n_glob, k_glob].
// ---------------------------------------------------------------------------
__global__ void pack_kernel(const float* __restrict__ W0,
                            const float* __restrict__ W1,
                            const float* __restrict__ W2,
                            const float* __restrict__ W3,
                            __bf16* __restrict__ P) {
  const int idx  = blockIdx.x * blockDim.x + threadIdx.x; // 0..262143
  const int mat  = idx >> 16;
  const int e    = idx & 65535;
  const int i    = e & 15;          // element within lane fragment
  const int lane = (e >> 4) & 31;
  const int kc   = (e >> 9) & 3;    // K chunk of 32
  const int jt   = e >> 11;         // N tile of 16 (0..31)
  const int ng   = jt * 16 + (lane & 15);
  const int kg   = kc * 32 + (i & 7) + 8 * (lane >> 4) + 16 * (i >> 3);
  const float* W = (mat == 0) ? W0 : (mat == 1) ? W1 : (mat == 2) ? W2 : W3;
  P[idx] = (__bf16)W[(size_t)ng * 128 + kg];
}

__global__ void zero_kernel(float* __restrict__ p, int n) {
  int i = blockIdx.x * blockDim.x + threadIdx.x;
  if (i < n) p[i] = 0.f;
}

// ---------------------------------------------------------------------------
// Per step: x_tilde = alpha * x_t  (written to output), + BN partial sums
// ---------------------------------------------------------------------------
__global__ void xtilde_bn_kernel(const float* __restrict__ X,
                                 const float* __restrict__ alpha,
                                 float* __restrict__ Xt_out,
                                 float* __restrict__ gsum,
                                 float* __restrict__ gsq,
                                 int t) {
  const int n    = threadIdx.x & 127;
  const int half = threadIdx.x >> 7;
  const int base = blockIdx.x * 32;
  float ps = 0.f, pq = 0.f;
  for (int r = half; r < 32; r += 2) {
    const size_t b   = base + r;
    const size_t off = b * (size_t)(Tsz * Nsz) + (size_t)t * Nsz + n;
    const float xt = alpha[b * Nsz + n] * X[off];
    Xt_out[off] = xt;
    ps += xt; pq += xt * xt;
  }
  __shared__ float s0[256], s1[256];
  s0[threadIdx.x] = ps; s1[threadIdx.x] = pq;
  __syncthreads();
  if (half == 0) {
    atomicAdd(&gsum[n], s0[n] + s0[n + 128]);
    atomicAdd(&gsq[n],  s1[n] + s1[n + 128]);
  }
}

__global__ void bn_finalize_kernel(float* __restrict__ gsum, float* __restrict__ gsq,
                                   const float* __restrict__ gamma,
                                   const float* __restrict__ beta,
                                   float* __restrict__ scale, float* __restrict__ shift) {
  const int n = threadIdx.x;
  const float inv = 1.0f / (float)Bsz;
  const float mean = gsum[n] * inv;
  const float var  = gsq[n] * inv - mean * mean;
  const float sc   = gamma[n] * rsqrtf(var + EPSf);
  scale[n] = sc;
  shift[n] = beta[n] - mean * sc;
  gsum[n] = 0.f; gsq[n] = 0.f;   // ready for next step
}

// ---------------------------------------------------------------------------
// Per step: fused 2-layer LSTM cell via bf16 WMMA (f32 accumulate).
// Block = 16 batch rows x full 512 gate columns; 8 waves, wave w owns
// jt in [4w, 4w+4).  LDS: sIn/sH bf16 operands, sG f32 gates (40KB total).
// ---------------------------------------------------------------------------
__global__ void __launch_bounds__(256)
lstm_step_kernel(const float* __restrict__ Xt_out,
                 const float* __restrict__ scale, const float* __restrict__ shift,
                 float* __restrict__ h0s, float* __restrict__ c0s,
                 float* __restrict__ h1s, float* __restrict__ c1s,
                 const __bf16* __restrict__ Pih0, const __bf16* __restrict__ Phh0,
                 const __bf16* __restrict__ Pih1, const __bf16* __restrict__ Phh1,
                 const float* __restrict__ bih0, const float* __restrict__ bhh0,
                 const float* __restrict__ bih1, const float* __restrict__ bhh1,
                 float* __restrict__ Xenc, int t)
{
  __shared__ __bf16 sIn[16 * 128];   // layer input (bf16)
  __shared__ __bf16 sH [16 * 128];   // recurrent input (bf16)
  __shared__ float  sG [16 * 512];   // gate accumulators (f32)

  const int tid  = threadIdx.x;
  const int lane = tid & 31;
  const int wave = tid >> 5;         // 0..7
  const int base = blockIdx.x * 16;  // batch row base
  const int m    = lane & 15;
  const int hi   = lane >> 4;

  // stage xb = BN(x_tilde) and h0_prev into LDS as bf16
  for (int idx = tid; idx < 16 * 128; idx += 256) {
    const int r = idx >> 7, n = idx & 127;
    const size_t b = base + r;
    const float xt = Xt_out[b * (size_t)(Tsz * Nsz) + (size_t)t * Nsz + n];
    sIn[idx] = (__bf16)(xt * scale[n] + shift[n]);
    sH[idx]  = (__bf16)h0s[b * Hsz + n];
  }
  __syncthreads();

  auto loadA = [&](const __bf16* src, v16bf* a) {
    // A fragment: lane<16 -> K in {0..7,16..23}, lane>=16 -> {8..15,24..31}
    const __bf16* row = src + m * 128;
    #pragma unroll
    for (int kc = 0; kc < 4; ++kc) {
      BF16x16 u;
      #pragma unroll
      for (int i = 0; i < 16; ++i)
        u.e[i] = row[kc * 32 + (i & 7) + 8 * hi + 16 * (i >> 3)];
      a[kc] = u.v;
    }
  };

  auto gemm = [&](const __bf16* Pi, const __bf16* Ph) {
    v16bf ax[4], ah[4];
    loadA(sIn, ax);
    loadA(sH,  ah);
    #pragma unroll
    for (int j = 0; j < 4; ++j) {
      const int jt = wave * 4 + j;
      v8f acc = {};
      #pragma unroll
      for (int kc = 0; kc < 4; ++kc) {
        v16bf bm = *(const v16bf*)(Pi + ((size_t)(jt * 4 + kc) * 32 + lane) * 16);
        acc = __builtin_amdgcn_wmma_f32_16x16x32_bf16(false, ax[kc], false, bm,
                                                      (short)0, acc, false, false);
      }
      #pragma unroll
      for (int kc = 0; kc < 4; ++kc) {
        v16bf bm = *(const v16bf*)(Ph + ((size_t)(jt * 4 + kc) * 32 + lane) * 16);
        acc = __builtin_amdgcn_wmma_f32_16x16x32_bf16(false, ah[kc], false, bm,
                                                      (short)0, acc, false, false);
      }
      // C layout: VGPR r -> M = r + 8*hi, N = m
      const int col = jt * 16 + m;
      #pragma unroll
      for (int r = 0; r < 8; ++r)
        sG[(8 * hi + r) * 512 + col] = acc[r];
    }
  };

  // ---- layer 0 ----
  gemm(Pih0, Phh0);
  __syncthreads();
  for (int idx = tid; idx < 16 * 128; idx += 256) {
    const int r = idx >> 7, hh = idx & 127;
    const size_t b = base + r;
    const float* g = sG + r * 512;
    const float gi = g[hh]       + bih0[hh]       + bhh0[hh];
    const float gf = g[128 + hh] + bih0[128 + hh] + bhh0[128 + hh];
    const float gg = g[256 + hh] + bih0[256 + hh] + bhh0[256 + hh];
    const float go = g[384 + hh] + bih0[384 + hh] + bhh0[384 + hh];
    const float cp = c0s[b * Hsz + hh];
    const float cn = sigmoidf(gf) * cp + sigmoidf(gi) * tanhf(gg);
    const float hn = sigmoidf(go) * tanhf(cn);
    c0s[b * Hsz + hh] = cn;
    h0s[b * Hsz + hh] = hn;
    sIn[idx] = (__bf16)hn;                       // layer-1 input
    sH[idx]  = (__bf16)h1s[b * Hsz + hh];        // layer-1 recurrent input
  }
  __syncthreads();

  // ---- layer 1 ----
  gemm(Pih1, Phh1);
  __syncthreads();
  for (int idx = tid; idx < 16 * 128; idx += 256) {
    const int r = idx >> 7, hh = idx & 127;
    const size_t b = base + r;
    const float* g = sG + r * 512;
    const float gi = g[hh]       + bih1[hh]       + bhh1[hh];
    const float gf = g[128 + hh] + bih1[128 + hh] + bhh1[128 + hh];
    const float gg = g[256 + hh] + bih1[256 + hh] + bhh1[256 + hh];
    const float go = g[384 + hh] + bih1[384 + hh] + bhh1[384 + hh];
    const float cp = c1s[b * Hsz + hh];
    const float cn = sigmoidf(gf) * cp + sigmoidf(gi) * tanhf(gg);
    const float hn = sigmoidf(go) * tanhf(cn);
    c1s[b * Hsz + hh] = cn;
    h1s[b * Hsz + hh] = hn;
    Xenc[b * (size_t)(Tsz * Hsz) + (size_t)t * Hsz + hh] = hn;
  }
}

// ---------------------------------------------------------------------------
extern "C" void kernel_launch(void* const* d_in, const int* in_sizes, int n_in,
                              void* d_out, int out_size, void* d_ws, size_t ws_size,
                              hipStream_t stream) {
  (void)in_sizes; (void)n_in; (void)out_size; (void)ws_size;

  const float* X      = (const float*)d_in[0];
  const float* attn_w = (const float*)d_in[1];
  const float* attn_b = (const float*)d_in[2];
  const float* gamma  = (const float*)d_in[3];
  const float* beta   = (const float*)d_in[4];
  const float* W_ih0  = (const float*)d_in[5];
  const float* W_hh0  = (const float*)d_in[6];
  const float* b_ih0  = (const float*)d_in[7];
  const float* b_hh0  = (const float*)d_in[8];
  const float* W_ih1  = (const float*)d_in[9];
  const float* W_hh1  = (const float*)d_in[10];
  const float* b_ih1  = (const float*)d_in[11];
  const float* b_hh1  = (const float*)d_in[12];

  char* ws = (char*)d_ws;
  size_t o = 0;
  float* alpha = (float*)(ws + o); o += (size_t)Bsz * Nsz * 4;
  float* h0s   = (float*)(ws + o); o += (size_t)Bsz * Hsz * 4;
  float* c0s   = (float*)(ws + o); o += (size_t)Bsz * Hsz * 4;
  float* h1s   = (float*)(ws + o); o += (size_t)Bsz * Hsz * 4;
  float* c1s   = (float*)(ws + o); o += (size_t)Bsz * Hsz * 4;
  float* gsum  = (float*)(ws + o); o += 128 * 4;
  float* gsq   = (float*)(ws + o); o += 128 * 4;
  float* scl   = (float*)(ws + o); o += 128 * 4;
  float* shf   = (float*)(ws + o); o += 128 * 4;
  __bf16* Pk   = (__bf16*)(ws + o);            // 4 x 65536 bf16 packed weights
  __bf16* Pih0 = Pk;
  __bf16* Phh0 = Pk + 65536;
  __bf16* Pih1 = Pk + 2 * 65536;
  __bf16* Phh1 = Pk + 3 * 65536;

  float* Xt_out = (float*)d_out;                          // (B,T,N)
  float* Xenc   = (float*)d_out + (size_t)Bsz * Tsz * Nsz; // (B,T,H)

  // one-time prep (re-run every call for determinism)
  alpha_kernel<<<Bsz, 128, 0, stream>>>(X, attn_w, attn_b, alpha);
  pack_kernel<<<1024, 256, 0, stream>>>(W_ih0, W_hh0, W_ih1, W_hh1, Pk);
  const int ZN = 4 * Bsz * Hsz + 256;  // states + gsum + gsq (contiguous)
  zero_kernel<<<(ZN + 255) / 256, 256, 0, stream>>>(h0s, ZN);

  for (int t = 0; t < Tsz; ++t) {
    xtilde_bn_kernel<<<Bsz / 32, 256, 0, stream>>>(X, alpha, Xt_out, gsum, gsq, t);
    bn_finalize_kernel<<<1, 128, 0, stream>>>(gsum, gsq, gamma, beta, scl, shf);
    lstm_step_kernel<<<Bsz / 16, 256, 0, stream>>>(Xt_out, scl, shf,
        h0s, c0s, h1s, c1s, Pih0, Phh0, Pih1, Phh1,
        b_ih0, b_hh0, b_ih1, b_hh1, Xenc, t);
  }
}
